// IFM_34076270526821
// MI455X (gfx1250) — matrix operand order repaired and checked by reading, not compiled, not run on hardware
//
#include <hip/hip_runtime.h>

// ---------------- problem constants (from reference) ----------------
#define Bc 4
#define Tc 8
#define Nc 2048
#define Mc 512      // N / SPATIAL_STRIDE
#define Kc 32
#define RADIUS2 0.25f
#define CIN 131
#define CINP 160    // padded to multiple of 32 for WMMA K-steps
#define H1 128
#define H2 128
#define H3 256

// workspace layout (bytes)
#define IDX_OFF    139264                      // after 69632 bf16 weights
#define IDXK_OFF   (139264 + Bc*Tc*Mc*4)

typedef __attribute__((ext_vector_type(16))) __bf16 v16bf;
typedef __attribute__((ext_vector_type(8)))  float  v8f;

static __device__ __forceinline__ unsigned short f2bfbits(float f) {
  unsigned u = __builtin_bit_cast(unsigned, f);
  unsigned r = u + 0x7FFFu + ((u >> 16) & 1u);   // round-to-nearest-even
  return (unsigned short)(r >> 16);
}

// ---------------- weight conversion: fp32 -> bf16 in WMMA B-fragment order ----------
// Fragment layout: frag[((nt*nKblk + kblk)*32 + lane)*16 + j] = W[k][n], where
//   k = kblk*32 + (lane>=16 ? 16 : 0) + j,  n = nt*16 + (lane&15)
// => each lane's 16 values for one v_wmma B operand are 32 contiguous bytes.
__global__ void wcvt_kernel(const float* __restrict__ w1, const float* __restrict__ w2,
                            const float* __restrict__ w3, unsigned short* __restrict__ wb) {
  const int S1 = CINP*H1, S2 = H1*H2, S3 = H2*H3;
  int gi = blockIdx.x * 256 + threadIdx.x;
  const float* w; int kdimP, kreal, ndim, off, i;
  if (gi < S1)            { w = w1; kdimP = CINP; kreal = CIN; ndim = H1; off = 0;     i = gi; }
  else if (gi < S1+S2)    { w = w2; kdimP = H1;   kreal = H1;  ndim = H2; off = S1;    i = gi - S1; }
  else if (gi < S1+S2+S3) { w = w3; kdimP = H2;   kreal = H2;  ndim = H3; off = S1+S2; i = gi - S1 - S2; }
  else return;
  int j    = i & 15;
  int lane = (i >> 4) & 31;
  int rest = i >> 9;
  int nKblk = kdimP >> 5;
  int kblk = rest % nKblk;
  int nt   = rest / nKblk;
  int k = (kblk << 5) + ((lane & 16) ? 16 : 0) + j;
  int n = (nt << 4) + (lane & 15);
  float v = (k < kreal) ? w[k * ndim + n] : 0.f;
  wb[off + i] = f2bfbits(v);
}

// ---------------- farthest point sampling: one block per (b,t) ----------------
__global__ void fps_kernel(const float* __restrict__ xyzs, int* __restrict__ idxOut,
                           float* __restrict__ anchorOut) {
  const int bt = blockIdx.x;      // 0..31
  const int tid = threadIdx.x;    // 0..255
  __shared__ float sPx[Nc], sPy[Nc], sPz[Nc], sDist[Nc];
  __shared__ float sRv[256]; __shared__ int sRi[256];
  __shared__ int   sIdx[Mc];
  __shared__ float sCur[3];

  const float* p = xyzs + (size_t)bt * Nc * 3;
  for (int j = tid; j < Nc; j += 256) {
    sPx[j] = p[j*3+0]; sPy[j] = p[j*3+1]; sPz[j] = p[j*3+2];
    sDist[j] = 1e10f;
  }
  if (tid == 0) { sIdx[0] = 0; sCur[0] = p[0]; sCur[1] = p[1]; sCur[2] = p[2]; }
  __syncthreads();

  for (int i = 0; i < Mc - 1; ++i) {
    float cx = sCur[0], cy = sCur[1], cz = sCur[2];
    float bv = -1.f; int bi = 0;
    for (int j = tid; j < Nc; j += 256) {
      float dx = sPx[j]-cx, dy = sPy[j]-cy, dz = sPz[j]-cz;
      float d = dx*dx + dy*dy + dz*dz;
      float od = sDist[j]; d = d < od ? d : od; sDist[j] = d;
      if (d > bv) { bv = d; bi = j; }   // strict >: first index wins (argmax tie rule)
    }
    sRv[tid] = bv; sRi[tid] = bi;
    __syncthreads();
    for (int s = 128; s > 0; s >>= 1) {
      if (tid < s) {
        float v2 = sRv[tid+s]; int i2 = sRi[tid+s];
        if (v2 > sRv[tid] || (v2 == sRv[tid] && i2 < sRi[tid])) { sRv[tid] = v2; sRi[tid] = i2; }
      }
      __syncthreads();
    }
    if (tid == 0) {
      int sel = sRi[0]; sIdx[i+1] = sel;
      sCur[0] = sPx[sel]; sCur[1] = sPy[sel]; sCur[2] = sPz[sel];
    }
    __syncthreads();
  }

  for (int m2 = tid; m2 < Mc; m2 += 256) {
    int sel = sIdx[m2];
    idxOut[bt*Mc + m2] = sel;
    anchorOut[(size_t)(bt*Mc + m2)*3 + 0] = sPx[sel];
    anchorOut[(size_t)(bt*Mc + m2)*3 + 1] = sPy[sel];
    anchorOut[(size_t)(bt*Mc + m2)*3 + 2] = sPz[sel];
  }
}

// ---------------- kNN in next frame: one block per anchor ----------------
__global__ void knn_kernel(const float* __restrict__ xyzs, const float* __restrict__ anchor,
                           int* __restrict__ idxkOut) {
  const int g = blockIdx.x;                 // 0..B*T*M-1
  const int bt = g / Mc;
  const int b = bt / Tc, t = bt - b*Tc;
  const int tn = (t + 1 < Tc) ? t + 1 : Tc - 1;
  const int tid = threadIdx.x;

  __shared__ float sDist[Nc];
  __shared__ float sRv[256]; __shared__ int sRi[256];
  __shared__ int sSel[Kc]; __shared__ float sSelD[Kc];

  const float* pn = xyzs + (size_t)(b*Tc + tn) * Nc * 3;
  float ax = anchor[(size_t)g*3+0], ay = anchor[(size_t)g*3+1], az = anchor[(size_t)g*3+2];
  for (int j = tid; j < Nc; j += 256) {
    float dx = pn[j*3]-ax, dy = pn[j*3+1]-ay, dz = pn[j*3+2]-az;
    sDist[j] = dx*dx + dy*dy + dz*dz;
  }
  __syncthreads();

  for (int p = 0; p < Kc; ++p) {
    float bv = 3.4e38f; int bi = 0;
    for (int j = tid; j < Nc; j += 256) {
      float v = sDist[j];
      if (v < bv) { bv = v; bi = j; }
    }
    sRv[tid] = bv; sRi[tid] = bi;
    __syncthreads();
    for (int s = 128; s > 0; s >>= 1) {
      if (tid < s) {
        float v2 = sRv[tid+s]; int i2 = sRi[tid+s];
        if (v2 < sRv[tid] || (v2 == sRv[tid] && i2 < sRi[tid])) { sRv[tid] = v2; sRi[tid] = i2; }
      }
      __syncthreads();
    }
    if (tid == 0) { sSel[p] = sRi[0]; sSelD[p] = sRv[0]; sDist[sRi[0]] = 3.9e38f; }
    __syncthreads();
  }

  if (tid < Kc) {
    int sel = (sSelD[tid] <= RADIUS2) ? sSel[tid] : sSel[0];   // radius fallback to NN
    idxkOut[(size_t)g*Kc + tid] = sel;
  }
}

// ---------------- dense layer tile engine (wave32 WMMA bf16) ----------------
// Each wave owns one 16-col n-tile; keeps two accumulators (rows 0-15, 16-31) so
// each pre-swizzled B fragment (2x b128 loads) feeds TWO back-to-back WMMAs.
// A fragments: 2x 16B LDS loads per M-tile per k-step.
template<int KDIM, int NDIM, int INSTRIDE, int OUTSTRIDE, bool OUTF32>
static __device__ __forceinline__ void mm_layer(
    const unsigned short* sIn, const unsigned short* __restrict__ gWfrag,
    const float* __restrict__ bias, unsigned short* sOutB, float* sOutF,
    int wave, int lane)
{
  constexpr int nNT   = NDIM >> 4;
  constexpr int nKblk = KDIM >> 5;
  const int l15 = lane & 15;
  const int hbB = (lane & 16) ? 32 : 0;   // byte offset selecting K-half of A row

  for (int nt = wave; nt < nNT; nt += 8) {
    v8f acc0 = {0.f,0.f,0.f,0.f,0.f,0.f,0.f,0.f};
    v8f acc1 = {0.f,0.f,0.f,0.f,0.f,0.f,0.f,0.f};
    #pragma unroll
    for (int kb = 0; kb < nKblk; ++kb) {
      union { v16bf v; uint4 q[2]; } a0, a1, bfr;
      // A k order used here (contiguous, same as B): lane<16 -> K 0-15; lane>=16 -> K 16-31
      const char* r0 = (const char*)(sIn + (size_t)l15        * INSTRIDE + (kb << 5));
      const char* r1 = (const char*)(sIn + (size_t)(l15 + 16) * INSTRIDE + (kb << 5));
      a0.q[0] = *(const uint4*)(r0 + hbB); a0.q[1] = *(const uint4*)(r0 + 16 + hbB);
      a1.q[0] = *(const uint4*)(r1 + hbB); a1.q[1] = *(const uint4*)(r1 + 16 + hbB);
      // B fragment: 32 contiguous bytes per lane (pre-swizzled by wcvt)
      const uint4* bp = (const uint4*)(gWfrag + (((size_t)nt * nKblk + kb) * 32 + lane) * 16);
      bfr.q[0] = bp[0]; bfr.q[1] = bp[1];
      acc0 = __builtin_amdgcn_wmma_f32_16x16x32_bf16(
          false, a0.v, false, bfr.v, (short)0, acc0, false, false);
      acc1 = __builtin_amdgcn_wmma_f32_16x16x32_bf16(
          false, a1.v, false, bfr.v, (short)0, acc1, false, false);
    }
    // C/D layout: VGPR r -> row r (lane<16) / 8+r (lane>=16); col = nt*16 + lane%16
    int col = (nt << 4) + l15;
    float bvv = bias[col];
    int rb = (lane & 16) ? 8 : 0;
    #pragma unroll
    for (int r = 0; r < 8; ++r) {
      float v0 = acc0[r] + bvv; v0 = v0 > 0.f ? v0 : 0.f;   // fused bias + ReLU
      float v1 = acc1[r] + bvv; v1 = v1 > 0.f ? v1 : 0.f;
      if constexpr (OUTF32) {
        sOutF[(rb + r)      * OUTSTRIDE + col] = v0;
        sOutF[(16 + rb + r) * OUTSTRIDE + col] = v1;
      } else {
        sOutB[(rb + r)      * OUTSTRIDE + col] = f2bfbits(v0);
        sOutB[(16 + rb + r) * OUTSTRIDE + col] = f2bfbits(v1);
      }
    }
  }
}

// ---------------- gather + 3-layer MLP + max-pool: one block (8 waves) per anchor ------
__global__ void mlp_kernel(const float* __restrict__ xyzs, const float* __restrict__ feats,
                           const unsigned short* __restrict__ wb,
                           const int* __restrict__ idx, const int* __restrict__ idxk,
                           const float* __restrict__ b1, const float* __restrict__ b2,
                           const float* __restrict__ b3,
                           const float* __restrict__ anchor, float* __restrict__ outNF)
{
  const int g = blockIdx.x;
  const int bt = g / Mc;
  const int b = bt / Tc, t = bt - b*Tc;
  const int tn = (t + 1 < Tc) ? t + 1 : Tc - 1;
  const int tid = threadIdx.x;
  const int wave = tid >> 5, lane = tid & 31;

  __shared__ alignas(16) unsigned short sH [32 * CINP];   // 10.0 KB  input 32x160 bf16
  __shared__ alignas(16) unsigned short sA1[32 * 136];    //  8.5 KB  act1 (128 + pad)
  __shared__ alignas(16) unsigned short sA2[32 * 136];    //  8.5 KB  act2
  __shared__ float sOut[32 * 264];                        // 33.0 KB  act3 32x256 f32
  __shared__ int sNbr[Kc];

  const unsigned short* w1b = wb;
  const unsigned short* w2b = wb + CINP*H1;
  const unsigned short* w3b = wb + CINP*H1 + H1*H2;
  __builtin_prefetch(w1b, 0, 2);   // global_prefetch_b8: warm weights into GL2
  __builtin_prefetch(w3b, 0, 2);

  const float* fA = feats + (size_t)(b*Tc + t ) * Nc * 64;
  const float* fN = feats + (size_t)(b*Tc + tn) * Nc * 64;
  const float* pN = xyzs  + (size_t)(b*Tc + tn) * Nc * 3;
  float ax = anchor[(size_t)g*3+0], ay = anchor[(size_t)g*3+1], az = anchor[(size_t)g*3+2];
  int aidx = idx[g];

  if (tid < Kc) sNbr[tid] = idxk[(size_t)g*Kc + tid];
  __syncthreads();

  // H = [ gf2 (64) | f1 (64) | disp (3) | pad (29) ] per neighbor row, bf16
  for (int ii = tid; ii < 32 * CINP; ii += 256) {
    int r = ii / CINP, c = ii - r * CINP;
    int nbr = sNbr[r];
    float v;
    if (c < 64)        v = fN[(size_t)nbr * 64 + c];
    else if (c < 128)  v = fA[(size_t)aidx * 64 + (c - 64)];
    else if (c == 128) v = pN[nbr*3+0] - ax;
    else if (c == 129) v = pN[nbr*3+1] - ay;
    else if (c == 130) v = pN[nbr*3+2] - az;
    else               v = 0.f;
    sH[ii] = f2bfbits(v);
  }
  __syncthreads();

  mm_layer<CINP, H1, CINP, 136, false>(sH , w1b, b1, sA1, nullptr, wave, lane);
  __syncthreads();
  mm_layer<H1,   H2, 136,  136, false>(sA1, w2b, b2, sA2, nullptr, wave, lane);
  __syncthreads();
  mm_layer<H2,   H3, 136,  264, true >(sA2, w3b, b3, nullptr, sOut, wave, lane);
  __syncthreads();

  // max over the 32 neighbors
  {
    int c = tid;                      // 256 threads == 256 output channels
    float mv = sOut[c];
    #pragma unroll 4
    for (int r = 1; r < 32; ++r) { float v = sOut[r*264 + c]; mv = v > mv ? v : mv; }
    outNF[(size_t)g * H3 + c] = mv;
  }
}

// ---------------- host-side launcher ----------------
extern "C" void kernel_launch(void* const* d_in, const int* in_sizes, int n_in,
                              void* d_out, int out_size, void* d_ws, size_t ws_size,
                              hipStream_t stream) {
  (void)in_sizes; (void)n_in; (void)out_size; (void)ws_size;
  const float* xyzs  = (const float*)d_in[0];
  const float* feats = (const float*)d_in[1];
  const float* w1 = (const float*)d_in[2];
  const float* b1 = (const float*)d_in[3];
  const float* w2 = (const float*)d_in[4];
  const float* b2 = (const float*)d_in[5];
  const float* w3 = (const float*)d_in[6];
  const float* b3 = (const float*)d_in[7];

  float* anchorOut = (float*)d_out;                          // [B,T,M,3]
  float* nfOut     = (float*)d_out + (size_t)Bc*Tc*Mc*3;     // [B,T,M,256]

  unsigned short* wbW = (unsigned short*)d_ws;
  int* idx  = (int*)((char*)d_ws + IDX_OFF);
  int* idxk = (int*)((char*)d_ws + IDXK_OFF);

  const int wtot = CINP*H1 + H1*H2 + H2*H3;
  wcvt_kernel<<<(wtot + 255)/256, 256, 0, stream>>>(w1, w2, w3, wbW);
  fps_kernel<<<Bc*Tc, 256, 0, stream>>>(xyzs, idx, anchorOut);
  knn_kernel<<<Bc*Tc*Mc, 256, 0, stream>>>(xyzs, anchorOut, idxk);
  mlp_kernel<<<Bc*Tc*Mc, 256, 0, stream>>>(xyzs, feats, wbW, idx, idxk,
                                           b1, b2, b3, anchorOut, nfOut);
}
